// Moe_4930622456030
// MI455X (gfx1250) — compile-verified, matrix operand order checked by reading, not run on hardware
//
#include <hip/hip_runtime.h>
#include <hip/hip_bf16.h>
#include <math.h>

typedef __attribute__((ext_vector_type(16))) __bf16          v16bf;
typedef __attribute__((ext_vector_type(16))) unsigned short  v16u;
typedef __attribute__((ext_vector_type(8)))  float           v8f;

#define T_TOK 32768
#define DIM_  512
#define HID_  2048
#define E_    8
#define NW_   ((size_t)E_ * HID_ * DIM_)   // elems per weight tensor

// ---- workspace layout (bytes) ----
#define WS_COUNTS  0
#define WS_CURSORS 32
#define WS_OFFSETS 64
#define WS_ROUTE   128
#define WS_WGT     (WS_ROUTE + T_TOK * 4)
#define WS_LIST    (WS_WGT + 2 * T_TOK * 4)
#define WS_WL      (WS_LIST + 2 * T_TOK * 4)
#define WS_W1B     (1u << 20)                       // bf16 w1, 16 MB
#define WS_W2B     (WS_W1B + (unsigned)(NW_ * 2))   // bf16 w2, 16 MB

// ---- ffn tiling ----
#define MT          64     // tokens per block
#define HB          64     // HID chunk
#define XS          528    // LDS stride for X (bf16 elems, mult of 16)
#define HS          80     // LDS stride for H
#define WS1         528    // LDS stride for staged w1 chunk
#define FFN_THREADS 512    // 16 waves: 4 (M) x 4 (N)
#define BLK_PER_E   (T_TOK / MT)

__device__ __forceinline__ unsigned short f2bf(float f) {
  unsigned u = __float_as_uint(f);
  u += 0x7FFFu + ((u >> 16) & 1u);          // round-to-nearest-even
  return (unsigned short)(u >> 16);
}

// raw 32-bit LDS offset for inline-asm async ops
__device__ __forceinline__ unsigned ldsOff(const void* p) {
  return (unsigned)(__SIZE_TYPE__)(const __attribute__((address_space(3))) void*)p;
}

// ---------------------------------------------------------------- zero
__global__ void moe_zero(float* __restrict__ out, int n, int* __restrict__ counts) {
  int i = blockIdx.x * 256 + threadIdx.x;
  if (i < n) out[i] = 0.0f;
  if (i < E_) counts[i] = 0;
}

// ---------------------------------------------------------------- weights fp32 -> bf16 (one pass)
__global__ __launch_bounds__(256) void moe_cvt(
    const float* __restrict__ w1, const float* __restrict__ w2,
    unsigned short* __restrict__ w1b, unsigned short* __restrict__ w2b) {
  size_t i4 = ((size_t)blockIdx.x * 256 + threadIdx.x) * 4;
  const float* src; unsigned short* dst; size_t off;
  if (i4 < NW_) { src = w1; dst = w1b; off = i4; }
  else          { src = w2; dst = w2b; off = i4 - NW_; }
  const float4 v = *(const float4*)(src + off);
  uint2 p;
  p.x = (unsigned)f2bf(v.x) | ((unsigned)f2bf(v.y) << 16);
  p.y = (unsigned)f2bf(v.z) | ((unsigned)f2bf(v.w) << 16);
  *(uint2*)(dst + off) = p;
}

// ---------------------------------------------------------------- gate (one wave32 per token)
__global__ __launch_bounds__(256) void moe_gate(
    const float* __restrict__ x, const float* __restrict__ gw,
    int* __restrict__ route, float* __restrict__ wgt, int* __restrict__ counts) {
  const int lane = threadIdx.x & 31;
  const int wid  = threadIdx.x >> 5;
  const int t    = blockIdx.x * 8 + wid;

  const float* xr = x + (size_t)t * DIM_ + lane * 16;
  float xv[16];
#pragma unroll
  for (int i = 0; i < 16; ++i) xv[i] = xr[i];

  float logit[E_];
#pragma unroll
  for (int e = 0; e < E_; ++e) {
    const float* g = gw + e * DIM_ + lane * 16;
    float s = 0.0f;
#pragma unroll
    for (int i = 0; i < 16; ++i) s += xv[i] * g[i];
#pragma unroll
    for (int off = 16; off > 0; off >>= 1) s += __shfl_xor(s, off, 32);
    logit[e] = s;
  }
  if (lane == 0) {
    int b0 = 0; float l0 = logit[0];
#pragma unroll
    for (int e = 1; e < E_; ++e) if (logit[e] > l0) { l0 = logit[e]; b0 = e; }
    int b1 = (b0 == 0) ? 1 : 0; float l1 = logit[b1];
#pragma unroll
    for (int e = 0; e < E_; ++e) if (e != b0 && logit[e] > l1) { l1 = logit[e]; b1 = e; }
    float e1  = __expf(l1 - l0);              // softmax over the top-2
    float inv = 1.0f / (1.0f + e1);
    route[t]       = b0 | (b1 << 8);
    wgt[2 * t]     = inv;
    wgt[2 * t + 1] = e1 * inv;
    atomicAdd(&counts[b0], 1);
    atomicAdd(&counts[b1], 1);
  }
}

// ---------------------------------------------------------------- scan (E=8)
__global__ void moe_scan(const int* __restrict__ counts, int* __restrict__ offsets,
                         int* __restrict__ cursors) {
  if (threadIdx.x == 0) {
    int acc = 0;
    for (int e = 0; e < E_; ++e) { offsets[e] = acc; acc += counts[e]; }
    offsets[E_] = acc;
  }
  if (threadIdx.x < E_) cursors[threadIdx.x] = 0;
}

// ---------------------------------------------------------------- scatter tokens to expert lists
__global__ void moe_scatter(const int* __restrict__ route, const float* __restrict__ wgt,
                            const int* __restrict__ offsets, int* __restrict__ cursors,
                            int* __restrict__ list, float* __restrict__ wl) {
  int t = blockIdx.x * 256 + threadIdx.x;
  if (t >= T_TOK) return;
  int r = route[t];
  int e0 = r & 255, e1 = (r >> 8) & 255;
  int p0 = atomicAdd(&cursors[e0], 1);
  list[offsets[e0] + p0] = t;
  wl  [offsets[e0] + p0] = wgt[2 * t];
  int p1 = atomicAdd(&cursors[e1], 1);
  list[offsets[e1] + p1] = t;
  wl  [offsets[e1] + p1] = wgt[2 * t + 1];
}

// ---------------------------------------------------------------- fused routed FFN (bf16 WMMA)
__global__ __launch_bounds__(FFN_THREADS) void moe_ffn(
    const float* __restrict__ x,
    const unsigned short* __restrict__ w1b, const unsigned short* __restrict__ w2b,
    const int* __restrict__ offsets, const int* __restrict__ list,
    const float* __restrict__ wl, float* __restrict__ out) {
  __shared__ alignas(32) unsigned short x_s [MT * XS];   // gathered X, bf16
  __shared__ alignas(32) unsigned short w1_s[HB * WS1];  // staged w1 chunk (async)
  __shared__ alignas(32) unsigned short h_s [MT * HS];   // hidden chunk, bf16
  __shared__ int   tok_s[MT];
  __shared__ float wt_s[MT];

  const int e    = blockIdx.x / BLK_PER_E;
  const int blk  = blockIdx.x % BLK_PER_E;
  const int beg  = offsets[e];
  const int end  = offsets[e + 1];
  const int base = beg + blk * MT;
  if (base >= end) return;                 // empty tail block for this expert

  const int tid  = threadIdx.x;
  const int lane = tid & 31;
  const int wid  = tid >> 5;
  const int mi   = wid & 3;                // M-wave (16 token rows)
  const int ni   = wid >> 2;               // N-wave
  const int lm   = lane & 15;
  const int lh   = lane >> 4;              // K-half selector

  if (tid < MT) {
    int g = base + tid;
    if (g < end) { tok_s[tid] = list[g]; wt_s[tid] = wl[g]; }
    else         { tok_s[tid] = list[base]; wt_s[tid] = 0.0f; }  // padded rows: weight 0
  }
  __syncthreads();

  // gather X rows -> LDS as bf16 (converted once per block)
  for (int i = tid; i < MT * (DIM_ / 4); i += FFN_THREADS) {
    int row = i / (DIM_ / 4);
    int c4  = (i % (DIM_ / 4)) * 4;
    const float4 v = *(const float4*)(x + (size_t)tok_s[row] * DIM_ + c4);
    uint2 p;
    p.x = (unsigned)f2bf(v.x) | ((unsigned)f2bf(v.y) << 16);
    p.y = (unsigned)f2bf(v.z) | ((unsigned)f2bf(v.w) << 16);
    *(uint2*)&x_s[row * XS + c4] = p;
  }

  const unsigned short* w1e = w1b + (size_t)e * HID_ * DIM_;
  const unsigned short* w2e = w2b + (size_t)e * DIM_ * HID_;

  v8f acc[8];                               // Y tile: 16 rows x 128 cols per wave
#pragma unroll
  for (int i = 0; i < 8; ++i) acc[i] = {};

  for (int hb = 0; hb < HID_; hb += HB) {
    __syncthreads();                        // x_s ready; h_s / w1_s free for reuse

    // ---- async-stage w1[hb:hb+HB, :] (bf16) into LDS: 8 x b128 per thread ----
    {
      const unsigned short* w1src = w1e + (size_t)hb * DIM_;
#pragma unroll
      for (int j = 0; j < 8; ++j) {
        int i   = tid + j * FFN_THREADS;    // chunk id, 16B each
        int row = i >> 6;                   // DIM_/8 = 64 chunks per row
        int c8  = (i & 63) * 8;
        unsigned long long ga = (unsigned long long)(const void*)(w1src + (size_t)row * DIM_ + c8);
        unsigned           lo = ldsOff(&w1_s[row * WS1 + c8]);
        asm volatile("global_load_async_to_lds_b128 %0, %1, off"
                     :: "v"(lo), "v"(ga) : "memory");
      }
      asm volatile("s_wait_asynccnt 0" ::: "memory");
    }
    // prefetch next w2 stripe -> global_prefetch_b8
    __builtin_prefetch((const void*)(w2e + (size_t)(128 * ni + lane) * HID_ + hb), 0, 1);
    __syncthreads();                        // all waves' async stages visible

    // ---- layer 1: h[64, HB] = relu(Xs @ w1chunk^T), wave = 16x16 tile ----
    v8f hacc = {};
#pragma unroll 4
    for (int k = 0; k < DIM_; k += 32) {
      v16u au = *(const v16u*)&x_s [(16 * mi + lm) * XS  + k + 16 * lh];
      v16u bu = *(const v16u*)&w1_s[(16 * ni + lm) * WS1 + k + 16 * lh];
      hacc = __builtin_amdgcn_wmma_f32_16x16x32_bf16(
          false, __builtin_bit_cast(v16bf, au),
          false, __builtin_bit_cast(v16bf, bu),
          (short)0, hacc, false, false);
    }
#pragma unroll
    for (int r = 0; r < 8; ++r) {           // relu -> bf16 -> LDS
      float v = hacc[r];
      v = v > 0.0f ? v : 0.0f;
      h_s[(16 * mi + r + 8 * lh) * HS + 16 * ni + lm] = f2bf(v);
    }
    __syncthreads();

    // ---- layer 2: Y += h @ w2[:, hb:hb+HB]^T (bf16 direct from global/L2) ----
#pragma unroll
    for (int k2 = 0; k2 < HB; k2 += 32) {
      v16u au = *(const v16u*)&h_s[(16 * mi + lm) * HS + k2 + 16 * lh];
      v16bf a = __builtin_bit_cast(v16bf, au);
#pragma unroll
      for (int nt = 0; nt < 8; ++nt) {
        int d = 128 * ni + 16 * nt + lm;                   // output column = w2 row
        v16u bu = *(const v16u*)(w2e + (size_t)d * HID_ + hb + k2 + 16 * lh);
        acc[nt] = __builtin_amdgcn_wmma_f32_16x16x32_bf16(
            false, a, false, __builtin_bit_cast(v16bf, bu),
            (short)0, acc[nt], false, false);
      }
    }
  }

  // ---- epilogue: scale by combine weight, scatter-add fp32 ----
#pragma unroll
  for (int nt = 0; nt < 8; ++nt) {
    int d = 128 * ni + 16 * nt + lm;
#pragma unroll
    for (int r = 0; r < 8; ++r) {
      int row = 16 * mi + r + 8 * lh;
      float v = acc[nt][r] * wt_s[row];
      atomicAdd(out + (size_t)tok_s[row] * DIM_ + d, v);
    }
  }
}

// ----------------------------------------------------------------
extern "C" void kernel_launch(void* const* d_in, const int* in_sizes, int n_in,
                              void* d_out, int out_size, void* d_ws, size_t ws_size,
                              hipStream_t stream) {
  (void)in_sizes; (void)n_in; (void)ws_size;
  const float* x  = (const float*)d_in[0];
  const float* gw = (const float*)d_in[1];
  const float* w1 = (const float*)d_in[2];
  const float* w2 = (const float*)d_in[3];
  float* out = (float*)d_out;

  char*  ws      = (char*)d_ws;
  int*   counts  = (int*)(ws + WS_COUNTS);
  int*   cursors = (int*)(ws + WS_CURSORS);
  int*   offsets = (int*)(ws + WS_OFFSETS);
  int*   route   = (int*)(ws + WS_ROUTE);
  float* wgt     = (float*)(ws + WS_WGT);
  int*   list    = (int*)(ws + WS_LIST);
  float* wl      = (float*)(ws + WS_WL);
  unsigned short* w1b = (unsigned short*)(ws + WS_W1B);
  unsigned short* w2b = (unsigned short*)(ws + WS_W2B);

  moe_zero   <<<(out_size + 255) / 256, 256, 0, stream>>>(out, out_size, counts);
  moe_cvt    <<<(int)(2 * NW_ / 4 / 256), 256, 0, stream>>>(w1, w2, w1b, w2b);
  moe_gate   <<<T_TOK / 8, 256, 0, stream>>>(x, gw, route, wgt, counts);
  moe_scan   <<<1, 32, 0, stream>>>(counts, offsets, cursors);
  moe_scatter<<<T_TOK / 256, 256, 0, stream>>>(route, wgt, offsets, cursors, list, wl);
  moe_ffn    <<<E_ * BLK_PER_E, FFN_THREADS, 0, stream>>>(x, w1b, w2b, offsets, list, wl, out);
}
